// DualSoftMaxMatcher_61014305407062
// MI455X (gfx1250) — matrix operand order repaired and checked by reading, not compiled, not run on hardware
//
#include <hip/hip_runtime.h>
#include <hip/hip_bf16.h>
#include <stdint.h>

#define BATCH 2
#define NROWS 8192
#define MCOLS 8192
#define DDIM  256

#define BM 128
#define BN 128
#define BK 32
#define LDT 40          // padded LDS row stride in bf16 elems (80B, 16B-aligned)
#define NSPLIT 8        // split-N factor for column reductions
#define KSTEPS (DDIM / BK)

typedef __attribute__((ext_vector_type(16))) __bf16 v16bf;
typedef __attribute__((ext_vector_type(8)))  float  v8f;
typedef __attribute__((ext_vector_type(4)))  int    v4i;

#define AS1 __attribute__((address_space(1)))
#define AS3 __attribute__((address_space(3)))

// CDNA5 async global->LDS copy path (ASYNCcnt-tracked), guarded so the file
// still compiles (host pass / older toolchains) via the sync staging fallback.
#if defined(__gfx1250__) && __has_builtin(__builtin_amdgcn_global_load_async_to_lds_b128)
#define USE_ASYNC_LDS 1
#else
#define USE_ASYNC_LDS 0
#endif

__device__ __forceinline__ void dsm_wait_async() {
#if USE_ASYNC_LDS
#if __has_builtin(__builtin_amdgcn_s_wait_asynccnt)
    __builtin_amdgcn_s_wait_asynccnt(0);
#else
    asm volatile("s_wait_asynccnt 0x0" ::: "memory");
#endif
#endif
}

union FragBf {
    v16bf v;
    uint4 q[2];
};

// ---------------------------------------------------------------------------
// K0: fp32 -> bf16 (round-to-nearest-even)
// ---------------------------------------------------------------------------
__global__ void __launch_bounds__(256)
dsm_cvt_bf16_kernel(const float* __restrict__ src, uint16_t* __restrict__ dst, int n) {
    int i = blockIdx.x * 256 + threadIdx.x;
    if (i < n) {
        uint32_t u = __float_as_uint(src[i]);
        uint32_t r = (u + 0x7FFFu + ((u >> 16) & 1u)) >> 16;
        dst[i] = (uint16_t)r;
    }
}

// ---------------------------------------------------------------------------
// K1: sim[b,n,m] = sum_k A[b,n,k]*B[b,m,k]   (bf16 WMMA, f32 accumulate)
// 128x128 tile per workgroup, 8 waves, each wave -> 32x64 (2x4 of 16x16).
// Double-buffered LDS; tiles filled with GLOBAL_LOAD_ASYNC_TO_LDS_B128 when
// available (ASYNCcnt), else global_load_b128 + ds_store_b128.
// ---------------------------------------------------------------------------
__global__ void __launch_bounds__(256)
dsm_gemm_bf16_kernel(const uint16_t* __restrict__ Abf,
                     const uint16_t* __restrict__ Bbf,
                     float* __restrict__ sim) {
    __shared__ uint16_t As[2][BM * LDT];
    __shared__ uint16_t Bs[2][BN * LDT];

    const int b    = blockIdx.z;
    const int tn   = blockIdx.y;          // tile along N (rows of sim)
    const int tm   = blockIdx.x;          // tile along M (cols of sim)
    const int tid  = threadIdx.x;
    const int lane = tid & 31;
    const int wave = tid >> 5;
    const int waveRow = (wave >> 1) * 32; // 0,32,64,96
    const int waveCol = (wave & 1) * 64;  // 0,64

    const uint16_t* __restrict__ Ag = Abf + ((size_t)b * NROWS + (size_t)tn * BM) * DDIM;
    const uint16_t* __restrict__ Bg = Bbf + ((size_t)b * MCOLS + (size_t)tm * BN) * DDIM;

    v8f zero = {};
    v8f acc[2][4];
#pragma unroll
    for (int i = 0; i < 2; ++i)
#pragma unroll
        for (int j = 0; j < 4; ++j) acc[i][j] = zero;

    const int ldRow = tid >> 1;           // 0..127
    const int ldK   = (tid & 1) * 16;     // 0 or 16 (each thread moves 32B per matrix)

    const int half = lane >> 4;           // 0/1 (half-wave select)
    const int l16  = lane & 15;

    // issue one K-step's tile fill (32B per thread per matrix) into buffer `buf`
    auto issue_tile = [&](int k0, int buf) {
        const uint16_t* ga = Ag + (size_t)ldRow * DDIM + k0 + ldK;
        const uint16_t* gb = Bg + (size_t)ldRow * DDIM + k0 + ldK;
        uint16_t* la = &As[buf][ldRow * LDT + ldK];
        uint16_t* lb = &Bs[buf][ldRow * LDT + ldK];
#if USE_ASYNC_LDS
        // IOFFSET is added to both the global and LDS addresses (ISA 10.7),
        // so offset:16 covers the second 16B half of the 32B chunk.
        __builtin_amdgcn_global_load_async_to_lds_b128((AS1 v4i*)(void*)ga, (AS3 v4i*)(void*)la, 0, 0);
        __builtin_amdgcn_global_load_async_to_lds_b128((AS1 v4i*)(void*)ga, (AS3 v4i*)(void*)la, 16, 0);
        __builtin_amdgcn_global_load_async_to_lds_b128((AS1 v4i*)(void*)gb, (AS3 v4i*)(void*)lb, 0, 0);
        __builtin_amdgcn_global_load_async_to_lds_b128((AS1 v4i*)(void*)gb, (AS3 v4i*)(void*)lb, 16, 0);
#else
        uint4 a0 = ((const uint4*)ga)[0], a1 = ((const uint4*)ga)[1];
        uint4 b0 = ((const uint4*)gb)[0], b1 = ((const uint4*)gb)[1];
        ((uint4*)la)[0] = a0; ((uint4*)la)[1] = a1;
        ((uint4*)lb)[0] = b0; ((uint4*)lb)[1] = b1;
#endif
    };

    issue_tile(0, 0);

    for (int ks = 0; ks < KSTEPS; ++ks) {
        const int buf = ks & 1;
        dsm_wait_async();      // own async copies into buf complete
        __syncthreads();       // all waves' copies complete / prior reads done
        if (ks + 1 < KSTEPS) issue_tile((ks + 1) * BK, buf ^ 1);

        // A fragments: 16x32 bf16, lane l<16: row M=l, K={0..7,16..23}; l>=16: K={8..15,24..31}
        FragBf af[2];
#pragma unroll
        for (int i = 0; i < 2; ++i) {
            int row = waveRow + i * 16 + l16;
            int kb  = half * 8;
            af[i].q[0] = *(const uint4*)(&As[buf][row * LDT + kb]);
            af[i].q[1] = *(const uint4*)(&As[buf][row * LDT + kb + 16]);
        }
        // B fragments: 32x16 bf16, lane l<16: col N=l, K={0..7,16..23}; l>=16: K={8..15,24..31}
        FragBf bfr[4];
#pragma unroll
        for (int j = 0; j < 4; ++j) {
            int col = waveCol + j * 16 + l16;
            int kb2 = half * 8;
            bfr[j].q[0] = *(const uint4*)(&Bs[buf][col * LDT + kb2]);
            bfr[j].q[1] = *(const uint4*)(&Bs[buf][col * LDT + kb2 + 16]);
        }
#pragma unroll
        for (int i = 0; i < 2; ++i)
#pragma unroll
            for (int j = 0; j < 4; ++j)
                acc[i][j] = __builtin_amdgcn_wmma_f32_16x16x32_bf16(
                    false, af[i].v, false, bfr[j].v, (short)0, acc[i][j], false, false);
    }

    // epilogue: C/D layout -> VGPR r holds M=r (lanes 0-15) and M=r+8 (lanes 16-31)
    const size_t simBase = (size_t)b * NROWS * MCOLS;
#pragma unroll
    for (int i = 0; i < 2; ++i) {
        int rowBase = tn * BM + waveRow + i * 16 + half * 8;
#pragma unroll
        for (int j = 0; j < 4; ++j) {
            int col = tm * BN + waveCol + j * 16 + l16;
#pragma unroll
            for (int r = 0; r < 8; ++r) {
                sim[simBase + (size_t)(rowBase + r) * MCOLS + col] = acc[i][j][r];
            }
        }
    }
}

// ---------------------------------------------------------------------------
// K2: row softmax stats (max, expsum) — one workgroup per (b,n) row,
// 32 floats per thread held in registers, LDS tree reduction
// ---------------------------------------------------------------------------
__global__ void __launch_bounds__(256)
dsm_row_stats_kernel(const float* __restrict__ sim,
                     float* __restrict__ rowmax, float* __restrict__ rowsum) {
    __shared__ float red[256];
    const int row = blockIdx.x;                    // 0..B*N-1
    const float* __restrict__ p = sim + (size_t)row * MCOLS;
    const int t = threadIdx.x;

    float x[32];
    float mx = -__builtin_inff();
#pragma unroll
    for (int i = 0; i < 32; ++i) { x[i] = p[t + 256 * i]; mx = fmaxf(mx, x[i]); }

    red[t] = mx; __syncthreads();
    for (int s = 128; s > 0; s >>= 1) {
        if (t < s) red[t] = fmaxf(red[t], red[t + s]);
        __syncthreads();
    }
    mx = red[0]; __syncthreads();

    float sum = 0.f;
#pragma unroll
    for (int i = 0; i < 32; ++i) sum += __expf(x[i] - mx);

    red[t] = sum; __syncthreads();
    for (int s = 128; s > 0; s >>= 1) {
        if (t < s) red[t] += red[t + s];
        __syncthreads();
    }
    if (t == 0) { rowmax[row] = mx; rowsum[row] = red[0]; }
}

// ---------------------------------------------------------------------------
// K3: column softmax stats — online softmax over split-N chunks, then merge
// ---------------------------------------------------------------------------
__global__ void __launch_bounds__(256)
dsm_col_stats_partial_kernel(const float* __restrict__ sim,
                             float* __restrict__ pm, float* __restrict__ ps) {
    const int m     = blockIdx.x * 256 + threadIdx.x;     // 0..M-1 (coalesced)
    const int split = blockIdx.y;                          // 0..NSPLIT-1
    const int b     = blockIdx.z;
    const float* __restrict__ p =
        sim + ((size_t)b * NROWS + (size_t)split * (NROWS / NSPLIT)) * MCOLS + m;

    float mx = -__builtin_inff();
    float s  = 0.f;
    for (int n = 0; n < NROWS / NSPLIT; ++n) {
        float x  = p[(size_t)n * MCOLS];
        float nm = fmaxf(mx, x);
        s = s * __expf(mx - nm) + __expf(x - nm);   // online softmax update
        mx = nm;
    }
    const int idx = split * (BATCH * MCOLS) + b * MCOLS + m;
    pm[idx] = mx; ps[idx] = s;
}

__global__ void __launch_bounds__(256)
dsm_col_stats_merge_kernel(const float* __restrict__ pm, const float* __restrict__ ps,
                           float* __restrict__ colmax, float* __restrict__ colsum) {
    const int i = blockIdx.x * 256 + threadIdx.x;          // 0..B*M-1
    if (i >= BATCH * MCOLS) return;
    float mx = -__builtin_inff();
#pragma unroll
    for (int s = 0; s < NSPLIT; ++s) mx = fmaxf(mx, pm[s * (BATCH * MCOLS) + i]);
    float sum = 0.f;
#pragma unroll
    for (int s = 0; s < NSPLIT; ++s)
        sum += ps[s * (BATCH * MCOLS) + i] * __expf(pm[s * (BATCH * MCOLS) + i] - mx);
    colmax[i] = mx; colsum[i] = sum;
}

// ---------------------------------------------------------------------------
// K4: P = exp(2x - rm - cm)/(rs*cs), written IN PLACE over sim (in d_out);
// also per-row max + argmax (best_B)
// ---------------------------------------------------------------------------
__global__ void __launch_bounds__(256)
dsm_pmat_kernel(float* __restrict__ P,
                const float* __restrict__ rowmax, const float* __restrict__ rowsum,
                const float* __restrict__ colmax, const float* __restrict__ colsum,
                float* __restrict__ rowPmax, int* __restrict__ bestB) {
    __shared__ float redv[256];
    __shared__ int   redi[256];
    const int row = blockIdx.x;                 // b*N + n
    const int b   = row / NROWS;
    const float rm = rowmax[row];
    const float rs = rowsum[row];
    float* __restrict__ p = P + (size_t)row * MCOLS;
    const float* __restrict__ cm = colmax + b * MCOLS;
    const float* __restrict__ cs = colsum + b * MCOLS;
    const int t = threadIdx.x;

    float best = -1.f;
    int   bi   = 0;
#pragma unroll
    for (int i = 0; i < 32; ++i) {
        int m = t + 256 * i;
        float x   = p[m];
        float val = __expf(2.f * x - rm - cm[m]) / (rs * cs[m]);
        p[m] = val;
        if (val > best) { best = val; bi = m; }
    }
    redv[t] = best; redi[t] = bi; __syncthreads();
    for (int s = 128; s > 0; s >>= 1) {
        if (t < s) {
            if (redv[t + s] > redv[t] ||
                (redv[t + s] == redv[t] && redi[t + s] < redi[t])) {
                redv[t] = redv[t + s]; redi[t] = redi[t + s];
            }
        }
        __syncthreads();
    }
    if (t == 0) { rowPmax[row] = redv[0]; bestB[row] = redi[0]; }
}

// ---------------------------------------------------------------------------
// K5: column max of P (split partials + merge)
// ---------------------------------------------------------------------------
__global__ void __launch_bounds__(256)
dsm_col_pmax_partial_kernel(const float* __restrict__ P, float* __restrict__ pq) {
    const int m     = blockIdx.x * 256 + threadIdx.x;
    const int split = blockIdx.y;
    const int b     = blockIdx.z;
    const float* __restrict__ p =
        P + ((size_t)b * NROWS + (size_t)split * (NROWS / NSPLIT)) * MCOLS + m;
    float mx = -__builtin_inff();
    for (int n = 0; n < NROWS / NSPLIT; ++n) mx = fmaxf(mx, p[(size_t)n * MCOLS]);
    pq[split * (BATCH * MCOLS) + b * MCOLS + m] = mx;
}

__global__ void __launch_bounds__(256)
dsm_col_pmax_merge_kernel(const float* __restrict__ pq, float* __restrict__ colPmax) {
    const int i = blockIdx.x * 256 + threadIdx.x;
    if (i >= BATCH * MCOLS) return;
    float mx = -__builtin_inff();
#pragma unroll
    for (int s = 0; s < NSPLIT; ++s) mx = fmaxf(mx, pq[s * (BATCH * MCOLS) + i]);
    colPmax[i] = mx;
}

// ---------------------------------------------------------------------------
// K6: valid / matches_A / matches_B
// ---------------------------------------------------------------------------
__global__ void __launch_bounds__(256)
dsm_finalize_kernel(const float* __restrict__ kpA, const float* __restrict__ kpB,
                    const float* __restrict__ rowPmax, const int* __restrict__ bestB,
                    const float* __restrict__ colPmax,
                    float* __restrict__ matchesA, float* __restrict__ matchesB,
                    float* __restrict__ valid) {
    const int row = blockIdx.x * 256 + threadIdx.x;        // 0..B*N-1
    if (row >= BATCH * NROWS) return;
    const int b  = row / NROWS;
    const int mb = bestB[row];
    const float pv  = rowPmax[row];
    const float cpm = colPmax[b * MCOLS + mb];
    matchesA[row * 2 + 0] = kpA[row * 2 + 0];
    matchesA[row * 2 + 1] = kpA[row * 2 + 1];
    matchesB[row * 2 + 0] = kpB[((size_t)b * MCOLS + mb) * 2 + 0];
    matchesB[row * 2 + 1] = kpB[((size_t)b * MCOLS + mb) * 2 + 1];
    valid[row] = (pv == cpm && pv > 0.0f) ? 1.0f : 0.0f;
}

// ---------------------------------------------------------------------------
extern "C" void kernel_launch(void* const* d_in, const int* in_sizes, int n_in,
                              void* d_out, int out_size, void* d_ws, size_t ws_size,
                              hipStream_t stream) {
    const float* kpA   = (const float*)d_in[0];   // [B,N,2]
    const float* descA = (const float*)d_in[1];   // [B,N,D]
    const float* kpB   = (const float*)d_in[2];   // [B,M,2]
    const float* descB = (const float*)d_in[3];   // [B,M,D]

    float* P        = (float*)d_out;
    float* matchesA = P + (size_t)BATCH * NROWS * MCOLS;
    float* matchesB = matchesA + (size_t)BATCH * NROWS * 2;
    float* valid    = matchesB + (size_t)BATCH * NROWS * 2;

    // bump allocator over d_ws (total ~20 MB)
    char*  ws  = (char*)d_ws;
    size_t off = 0;
    auto alloc = [&](size_t bytes) -> char* {
        char* p = ws + off;
        off = (off + bytes + 255) & ~(size_t)255;
        return p;
    };
    uint16_t* Abf    = (uint16_t*)alloc((size_t)BATCH * NROWS * DDIM * 2);
    uint16_t* Bbf    = (uint16_t*)alloc((size_t)BATCH * MCOLS * DDIM * 2);
    float* rowmax    = (float*)alloc((size_t)BATCH * NROWS * 4);
    float* rowsum    = (float*)alloc((size_t)BATCH * NROWS * 4);
    float* colmax    = (float*)alloc((size_t)BATCH * MCOLS * 4);
    float* colsum    = (float*)alloc((size_t)BATCH * MCOLS * 4);
    float* pm        = (float*)alloc((size_t)NSPLIT * BATCH * MCOLS * 4);
    float* ps        = (float*)alloc((size_t)NSPLIT * BATCH * MCOLS * 4);
    float* pq        = (float*)alloc((size_t)NSPLIT * BATCH * MCOLS * 4);
    float* rowPmax   = (float*)alloc((size_t)BATCH * NROWS * 4);
    int*   bestB     = (int*)  alloc((size_t)BATCH * NROWS * 4);
    float* colPmax   = (float*)alloc((size_t)BATCH * MCOLS * 4);
    (void)ws_size; (void)in_sizes; (void)n_in; (void)out_size;

    // K0: fp32 -> bf16 descriptors (16 MB total; fits L2 thereafter)
    {
        int n = BATCH * NROWS * DDIM;
        dsm_cvt_bf16_kernel<<<(n + 255) / 256, 256, 0, stream>>>(descA, Abf, n);
        int m = BATCH * MCOLS * DDIM;
        dsm_cvt_bf16_kernel<<<(m + 255) / 256, 256, 0, stream>>>(descB, Bbf, m);
    }

    // K1: bf16 WMMA GEMM -> sim staged in d_out's P region
    dsm_gemm_bf16_kernel<<<dim3(MCOLS / BN, NROWS / BM, BATCH), 256, 0, stream>>>(Abf, Bbf, P);

    // K2: row softmax stats
    dsm_row_stats_kernel<<<BATCH * NROWS, 256, 0, stream>>>(P, rowmax, rowsum);

    // K3: column softmax stats (online, split-N) + merge
    dsm_col_stats_partial_kernel<<<dim3(MCOLS / 256, NSPLIT, BATCH), 256, 0, stream>>>(P, pm, ps);
    dsm_col_stats_merge_kernel<<<(BATCH * MCOLS + 255) / 256, 256, 0, stream>>>(pm, ps, colmax, colsum);

    // K4: P in place + row max/argmax
    dsm_pmat_kernel<<<BATCH * NROWS, 256, 0, stream>>>(P, rowmax, rowsum, colmax, colsum,
                                                       rowPmax, bestB);

    // K5: column max of P + merge
    dsm_col_pmax_partial_kernel<<<dim3(MCOLS / 256, NSPLIT, BATCH), 256, 0, stream>>>(P, pq);
    dsm_col_pmax_merge_kernel<<<(BATCH * MCOLS + 255) / 256, 256, 0, stream>>>(pq, colPmax);

    // K6: matches + valid
    dsm_finalize_kernel<<<(BATCH * NROWS + 255) / 256, 256, 0, stream>>>(
        kpA, kpB, rowPmax, bestB, colPmax, matchesA, matchesB, valid);
}
